// NodeLevelGNNAnomalyDetector_21603685499208
// MI455X (gfx1250) — matrix operand order repaired and checked by reading, not compiled, not run on hardware
//
#include <hip/hip_runtime.h>
#include <hip/hip_bf16.h>

typedef __attribute__((ext_vector_type(16))) _Float16 v16h;
typedef __attribute__((ext_vector_type(8)))  float    v8f;

#define D_IN   128
#define D_H    64
#define HEADS  4
#define GDIM   (HEADS * D_H)   // 256
#define NEG_SLOPE 0.2f

// ---------------------------------------------------------------------------
// WMMA GEMM: out[M,C] = act(A[M,K] @ W[K,C] + bias)
// fp32 inputs converted to f16 fragments, fp32 accumulate via
// v_wmma_f32_16x16x32_f16. One wave per 16x16 output tile.
// K, C compile-time: k-loop fully unrolled, constant addressing.
// Requires: M % 16 == 0, K % 32 == 0. C arbitrary (stores guarded; B-column
// clamped for loads -- garbage only lands in never-stored columns since each
// output column depends solely on its own B column).
// Fragment layouts used here (wave32; A/B k is contiguous, not the ISA table order):
//   A 16x32 f16 : lane L holds row M=L&15; lanes 0-15 K=0..15,
//                 lanes 16-31 K=16..31
//   B 32x16 f16 : lane L holds col N=L&15; lanes 0-15 K=0..15, lanes 16-31 K=16..31
//   C/D 16x16 f32: vgpr r = rows {r, r+8}; lane&15 = col
// ---------------------------------------------------------------------------
template <int ACT, int K, int C>
__global__ void wmma_gemm_bias(const float* __restrict__ A,
                               const float* __restrict__ W,
                               const float* __restrict__ bias,
                               float* __restrict__ out,
                               int M) {
  constexpr int CT = (C + 15) >> 4;   // column tiles
  const long long gtid = (long long)blockIdx.x * blockDim.x + threadIdx.x;
  const int wave = (int)(gtid >> 5);
  const int lane = threadIdx.x & 31;
  const int row0 = (wave / CT) << 4;
  const int col0 = (wave % CT) << 4;
  if (row0 >= M) return;

  const int mrow    = row0 + (lane & 15);          // A row handled by this lane
  const int ncol    = col0 + (lane & 15);          // out column handled by this lane
  const bool colok  = (ncol < C);
  const int ncol_ld = colok ? ncol : (C - 1);      // clamped: loads always in-bounds
  const int kbaseA  = (lane < 16) ? 0 : 16;
  const int kbaseB  = (lane < 16) ? 0 : 16;
  const float bv    = (bias && colok) ? bias[ncol] : 0.0f;

  const float* __restrict__ arow = A + (size_t)mrow * K;
  const float* __restrict__ wcol = W + ncol_ld;

  v8f acc = {};
#pragma unroll
  for (int k0 = 0; k0 < K; k0 += 32) {
    v16h a, b;
#pragma unroll
    for (int i = 0; i < 8; ++i) {
      a[i]     = (_Float16)arow[k0 + kbaseA + i];
      a[8 + i] = (_Float16)arow[k0 + kbaseA + 8 + i];
    }
#pragma unroll
    for (int i = 0; i < 16; ++i) {
      b[i] = (_Float16)wcol[(size_t)(k0 + kbaseB + i) * C];
    }
    acc = __builtin_amdgcn_wmma_f32_16x16x32_f16(
        /*neg_a=*/false, a, /*neg_b=*/false, b,
        /*c_mod=*/(short)0, acc, /*reuse_a=*/false, /*reuse_b=*/false);
  }

  if (colok) {
    const int mhi = (lane < 16) ? 0 : 8;
#pragma unroll
    for (int r = 0; r < 8; ++r) {
      const int m = row0 + r + mhi;
      float v = acc[r] + bv;
      if (ACT) v = fmaxf(v, 0.0f);
      out[(size_t)m * C + ncol] = v;
    }
  }
}

// ---------------------------------------------------------------------------
// Degree / normalization
// ---------------------------------------------------------------------------
__global__ void k_init_const(float* __restrict__ p, float v, long long n) {
  long long i = (long long)blockIdx.x * blockDim.x + threadIdx.x;
  if (i < n) p[i] = v;
}

__global__ void k_deg_count(const int* __restrict__ dst, float* __restrict__ deg, int E) {
  long long i = (long long)blockIdx.x * blockDim.x + threadIdx.x;
  if (i < E) atomicAdd(&deg[dst[i]], 1.0f);
}

__global__ void k_deg_finalize(float* __restrict__ deg_to_dis, float* __restrict__ dinv, int N) {
  long long i = (long long)blockIdx.x * blockDim.x + threadIdx.x;
  if (i < N) {
    float d = deg_to_dis[i];
    deg_to_dis[i] = rsqrtf(d);
    dinv[i] = 1.0f / d;
  }
}

// ---------------------------------------------------------------------------
// GCN: agg = hw*dinv + b (self loop)  ->  += norm-weighted neighbors -> relu(+res)
// ---------------------------------------------------------------------------
__global__ void k_gcn_self(const float* __restrict__ hw, const float* __restrict__ dinv,
                           const float* __restrict__ b, float* __restrict__ agg, int N) {
  long long i = (long long)blockIdx.x * blockDim.x + threadIdx.x;
  if (i < (long long)N * D_H) {
    int n = (int)(i >> 6), c = (int)(i & 63);
    agg[i] = hw[i] * dinv[n] + b[c];
  }
}

__global__ void k_gcn_edge(const int* __restrict__ src, const int* __restrict__ dst,
                           const float* __restrict__ hw, const float* __restrict__ dis,
                           float* __restrict__ agg, int E) {
  long long i = (long long)blockIdx.x * blockDim.x + threadIdx.x;
  if (i < (long long)E * D_H) {
    int e = (int)(i >> 6), c = (int)(i & 63);
    int s = src[e], d = dst[e];
    atomicAdd(&agg[(size_t)d * D_H + c], hw[(size_t)s * D_H + c] * dis[s] * dis[d]);
  }
}

__global__ void k_gcn_act(const float* __restrict__ agg, const float* __restrict__ hprev,
                          float* __restrict__ hout, long long total) {
  long long i = (long long)blockIdx.x * blockDim.x + threadIdx.x;
  if (i < total) {
    float v = fmaxf(agg[i], 0.0f);
    hout[i] = hprev ? (hprev[i] + v) : v;
  }
}

// ---------------------------------------------------------------------------
// GAT
// ---------------------------------------------------------------------------
__device__ inline void atomicMaxFloat(float* addr, float val) {
  if (val >= 0.0f) atomicMax((int*)addr, __float_as_int(val));
  else             atomicMin((unsigned int*)addr, __float_as_uint(val));
}

// per (n, h): attention linear terms; also init m/den
__global__ void k_gat_alpha(const float* __restrict__ g, const float* __restrict__ a_src,
                            const float* __restrict__ a_dst, float* __restrict__ als,
                            float* __restrict__ ald, float* __restrict__ m,
                            float* __restrict__ den, int N) {
  long long i = (long long)blockIdx.x * blockDim.x + threadIdx.x;
  if (i < (long long)N * HEADS) {
    int n = (int)(i >> 2), h = (int)(i & 3);
    const float* gp = g + (size_t)n * GDIM + h * D_H;
    const float* as = a_src + h * D_H;
    const float* ad = a_dst + h * D_H;
    float s = 0.f, d = 0.f;
#pragma unroll 8
    for (int c = 0; c < D_H; ++c) { float gv = gp[c]; s += gv * as[c]; d += gv * ad[c]; }
    als[i] = s; ald[i] = d;
    m[i] = -3.0e38f; den[i] = 0.0f;
  }
}

__global__ void k_gat_edge_logit(const int* __restrict__ src, const int* __restrict__ dst,
                                 const float* __restrict__ als, const float* __restrict__ ald,
                                 float* __restrict__ ebuf, float* __restrict__ m, int E) {
  long long i = (long long)blockIdx.x * blockDim.x + threadIdx.x;
  if (i < (long long)E * HEADS) {
    int e = (int)(i >> 2), h = (int)(i & 3);
    float v = als[(size_t)src[e] * HEADS + h] + ald[(size_t)dst[e] * HEADS + h];
    v = (v > 0.f) ? v : NEG_SLOPE * v;
    ebuf[i] = v;
    atomicMaxFloat(&m[(size_t)dst[e] * HEADS + h], v);
  }
}

__global__ void k_gat_loop_logit(const float* __restrict__ als, const float* __restrict__ ald,
                                 float* __restrict__ eloop, float* __restrict__ m, int N) {
  long long i = (long long)blockIdx.x * blockDim.x + threadIdx.x;
  if (i < (long long)N * HEADS) {
    float v = als[i] + ald[i];
    v = (v > 0.f) ? v : NEG_SLOPE * v;
    eloop[i] = v;
    atomicMaxFloat(&m[i], v);
  }
}

__global__ void k_gat_edge_exp(const int* __restrict__ dst, const float* __restrict__ m,
                               float* __restrict__ ebuf, float* __restrict__ den, int E) {
  long long i = (long long)blockIdx.x * blockDim.x + threadIdx.x;
  if (i < (long long)E * HEADS) {
    int e = (int)(i >> 2), h = (int)(i & 3);
    float ex = __expf(ebuf[i] - m[(size_t)dst[e] * HEADS + h]);
    ebuf[i] = ex;
    atomicAdd(&den[(size_t)dst[e] * HEADS + h], ex);
  }
}

__global__ void k_gat_loop_exp(const float* __restrict__ m, float* __restrict__ eloop,
                               float* __restrict__ den, int N) {
  long long i = (long long)blockIdx.x * blockDim.x + threadIdx.x;
  if (i < (long long)N * HEADS) {
    float ex = __expf(eloop[i] - m[i]);
    eloop[i] = ex;
    atomicAdd(&den[i], ex);
  }
}

__global__ void k_gat_edge_agg(const int* __restrict__ src, const int* __restrict__ dst,
                               const float* __restrict__ ebuf, const float* __restrict__ den,
                               const float* __restrict__ g, float* __restrict__ oagg, int E) {
  long long i = (long long)blockIdx.x * blockDim.x + threadIdx.x;
  if (i < (long long)E * GDIM) {
    int e = (int)(i >> 8);
    int hc = (int)(i & 255);
    int h = hc >> 6;
    int s = src[e], d = dst[e];
    float alpha = ebuf[(size_t)e * HEADS + h] / den[(size_t)d * HEADS + h];
    atomicAdd(&oagg[(size_t)d * GDIM + hc], g[(size_t)s * GDIM + hc] * alpha);
  }
}

// add self-loop contribution, mean over heads, + bias -> h_out [N, 64]
__global__ void k_gat_finish(const float* __restrict__ eloop, const float* __restrict__ den,
                             const float* __restrict__ g, const float* __restrict__ bg,
                             const float* __restrict__ oagg, float* __restrict__ hout, int N) {
  long long i = (long long)blockIdx.x * blockDim.x + threadIdx.x;
  if (i < (long long)N * D_H) {
    int n = (int)(i >> 6), c = (int)(i & 63);
    float sum = 0.f;
#pragma unroll
    for (int h = 0; h < HEADS; ++h) {
      float alpha = eloop[(size_t)n * HEADS + h] / den[(size_t)n * HEADS + h];
      float gv = g[(size_t)n * GDIM + h * D_H + c];
      sum += oagg[(size_t)n * GDIM + h * D_H + c] + gv * alpha;
    }
    hout[(size_t)n * D_H + c] = sum * (1.0f / HEADS) + bg[c];
  }
}

// ---------------------------------------------------------------------------
// Launch
// ---------------------------------------------------------------------------
static inline int nblk(long long threads, int bs) { return (int)((threads + bs - 1) / bs); }

static inline int gemm_blocks(int M, int C, int bs) {
  long long waves = (long long)(M / 16) * ((C + 15) / 16);
  return nblk(waves * 32, bs);
}

extern "C" void kernel_launch(void* const* d_in, const int* in_sizes, int n_in,
                              void* d_out, int out_size, void* d_ws, size_t ws_size,
                              hipStream_t stream) {
  const int N = in_sizes[0] / D_IN;
  const int E = in_sizes[1];

  const float* x     = (const float*)d_in[0];
  const int*   src   = (const int*)  d_in[1];
  const int*   dst   = (const int*)  d_in[2];
  const float* W0    = (const float*)d_in[3];
  const float* b0    = (const float*)d_in[4];
  const float* W1    = (const float*)d_in[5];
  const float* b1    = (const float*)d_in[6];
  const float* W2    = (const float*)d_in[7];
  const float* b2    = (const float*)d_in[8];
  const float* Wg    = (const float*)d_in[9];
  const float* a_src = (const float*)d_in[10];
  const float* a_dst = (const float*)d_in[11];
  const float* bg    = (const float*)d_in[12];
  const float* Wc1   = (const float*)d_in[13];
  const float* bc1   = (const float*)d_in[14];
  const float* Wc2   = (const float*)d_in[15];
  const float* bc2   = (const float*)d_in[16];
  const float* Wr1   = (const float*)d_in[17];
  const float* br1   = (const float*)d_in[18];
  const float* Wr2   = (const float*)d_in[19];
  const float* br2   = (const float*)d_in[20];

  float* outp = (float*)d_out;
  float* out_cls = outp;                          // [N, 3]
  float* out_rec = outp + (size_t)N * 3;          // [N, 128]
  float* out_h   = outp + (size_t)N * (3 + D_IN); // [N, 64]  (final GAT hidden)

  // ---- workspace partition (floats) ----
  float* ws = (float*)d_ws;
  size_t off = 0;
  auto alloc = [&](size_t n) { float* p = ws + off; off += n; return p; };
  float* dis   = alloc(N);                     // deg -> deg^-1/2
  float* dinv  = alloc(N);
  float* h     = alloc((size_t)N * D_H);
  float* h2    = alloc((size_t)N * D_H);       // agg buffer / reused as r1
  float* hw    = alloc((size_t)N * D_H);       // projection / reused as t1
  float* g     = alloc((size_t)N * GDIM);
  float* als   = alloc((size_t)N * HEADS);
  float* ald   = alloc((size_t)N * HEADS);
  float* mbuf  = alloc((size_t)N * HEADS);
  float* den   = alloc((size_t)N * HEADS);
  float* eloop = alloc((size_t)N * HEADS);
  float* oagg  = alloc((size_t)N * GDIM);
  float* ebuf  = alloc((size_t)E * HEADS);
  (void)ws_size; (void)n_in; (void)out_size;

  const int BS = 256;

  // ---- degrees ----
  k_init_const<<<nblk(N, BS), BS, 0, stream>>>(dis, 1.0f, N);     // self loop
  k_deg_count<<<nblk(E, BS), BS, 0, stream>>>(dst, dis, E);
  k_deg_finalize<<<nblk(N, BS), BS, 0, stream>>>(dis, dinv, N);

  // ---- GCN layer 0: 128 -> 64, h = relu(agg) ----
  wmma_gemm_bias<0, D_IN, D_H><<<gemm_blocks(N, D_H, BS), BS, 0, stream>>>(x, W0, nullptr, hw, N);
  k_gcn_self<<<nblk((long long)N * D_H, BS), BS, 0, stream>>>(hw, dinv, b0, h2, N);
  k_gcn_edge<<<nblk((long long)E * D_H, BS), BS, 0, stream>>>(src, dst, hw, dis, h2, E);
  k_gcn_act<<<nblk((long long)N * D_H, BS), BS, 0, stream>>>(h2, nullptr, h, (long long)N * D_H);

  // ---- GCN layers 1,2 with residual ----
  const float* Ws[2] = {W1, W2};
  const float* bs_[2] = {b1, b2};
  for (int l = 0; l < 2; ++l) {
    wmma_gemm_bias<0, D_H, D_H><<<gemm_blocks(N, D_H, BS), BS, 0, stream>>>(h, Ws[l], nullptr, hw, N);
    k_gcn_self<<<nblk((long long)N * D_H, BS), BS, 0, stream>>>(hw, dinv, bs_[l], h2, N);
    k_gcn_edge<<<nblk((long long)E * D_H, BS), BS, 0, stream>>>(src, dst, hw, dis, h2, E);
    k_gcn_act<<<nblk((long long)N * D_H, BS), BS, 0, stream>>>(h2, h, h, (long long)N * D_H);
  }

  // ---- GAT ----
  wmma_gemm_bias<0, D_H, GDIM><<<gemm_blocks(N, GDIM, BS), BS, 0, stream>>>(h, Wg, nullptr, g, N);
  k_gat_alpha<<<nblk((long long)N * HEADS, BS), BS, 0, stream>>>(g, a_src, a_dst, als, ald, mbuf, den, N);
  k_gat_edge_logit<<<nblk((long long)E * HEADS, BS), BS, 0, stream>>>(src, dst, als, ald, ebuf, mbuf, E);
  k_gat_loop_logit<<<nblk((long long)N * HEADS, BS), BS, 0, stream>>>(als, ald, eloop, mbuf, N);
  k_gat_edge_exp<<<nblk((long long)E * HEADS, BS), BS, 0, stream>>>(dst, mbuf, ebuf, den, E);
  k_gat_loop_exp<<<nblk((long long)N * HEADS, BS), BS, 0, stream>>>(mbuf, eloop, den, N);
  k_init_const<<<nblk((long long)N * GDIM, BS), BS, 0, stream>>>(oagg, 0.0f, (long long)N * GDIM);
  k_gat_edge_agg<<<nblk((long long)E * GDIM, BS), BS, 0, stream>>>(src, dst, ebuf, den, g, oagg, E);
  k_gat_finish<<<nblk((long long)N * D_H, BS), BS, 0, stream>>>(eloop, den, g, bg, oagg, out_h, N);

  // ---- classifier head: 64 -> 32 -> 3 ----
  wmma_gemm_bias<1, D_H, 32><<<gemm_blocks(N, 32, BS), BS, 0, stream>>>(out_h, Wc1, bc1, hw, N);
  wmma_gemm_bias<0, 32, 3><<<gemm_blocks(N, 3, BS), BS, 0, stream>>>(hw, Wc2, bc2, out_cls, N);

  // ---- reconstructor head: 64 -> 64 -> 128 ----
  wmma_gemm_bias<1, D_H, D_H><<<gemm_blocks(N, D_H, BS), BS, 0, stream>>>(out_h, Wr1, br1, h2, N);
  wmma_gemm_bias<0, D_H, D_IN><<<gemm_blocks(N, D_IN, BS), BS, 0, stream>>>(h2, Wr2, br2, out_rec, N);
}
